// GlobalBUAttentionGeneral_60181081752391
// MI455X (gfx1250) — matrix operand order repaired and checked by reading, not compiled, not run on hardware
//
#include <hip/hip_runtime.h>
#include <hip/hip_bf16.h>
#include <math.h>

#define BB 32
#define DF 256
#define QK 4096
#define LL 64
#define CDF 300

#define TROW 264            // bf16 tile row stride (256 + 8 pad, 16B aligned)
#define AROW 68             // f32 attn row stride (64 + 4 pad)

typedef __attribute__((ext_vector_type(16))) __bf16 v16bf;
typedef __attribute__((ext_vector_type(8)))  __bf16 v8bf;
typedef __attribute__((ext_vector_type(8)))  float  v8f;

__device__ __forceinline__ v16bf cat8(v8bf lo, v8bf hi) {
    return __builtin_shufflevector(lo, hi, 0,1,2,3,4,5,6,7,8,9,10,11,12,13,14,15);
}

__device__ __forceinline__ v8f wmma_bf16(v16bf a, v16bf b, v8f c) {
    // D = A(16x32 bf16) * B(32x16 bf16) + C(16x16 f32)
    return __builtin_amdgcn_wmma_f32_16x16x32_bf16(false, a, false, b, (short)0, c, false, false);
}

// ---------------------------------------------------------------------------
// K1: norm_c[b,l] = ||context1[b,:,l]||  and ctx1T[b,l,d] = bf16(context1[b,d,l])
// ---------------------------------------------------------------------------
__global__ __launch_bounds__(256)
void k_prep_ctx(const float* __restrict__ ctx1, __bf16* __restrict__ ctx1T,
                float* __restrict__ norm_c) {
    __shared__ float red[256];
    const int b = blockIdx.x;
    const int tid = threadIdx.x;
    const int l = tid & 63;
    const int g = tid >> 6;
    const float* base = ctx1 + (size_t)b * DF * LL;

    float s = 0.f;
    for (int d = g * 64; d < g * 64 + 64; ++d) {
        const float x = base[(size_t)d * LL + l];
        s += x * x;
    }
    red[tid] = s;
    __syncthreads();
    if (tid < 64)
        norm_c[b * LL + tid] = sqrtf(red[tid] + red[tid + 64] + red[tid + 128] + red[tid + 192]);
    __syncthreads();
    for (int idx = tid; idx < DF * LL; idx += 256) {
        const int d  = idx & 255;
        const int l2 = idx >> 8;
        ctx1T[((size_t)b * LL + l2) * DF + d] = (__bf16)base[(size_t)d * LL + l2];
    }
}

// ---------------------------------------------------------------------------
// K2: sourceT[b,d,l] = sum_c W[d,c] * ctx2[b,c,l]  (bf16 out, K=300)
// ---------------------------------------------------------------------------
__global__ __launch_bounds__(256)
void k_prep_source(const float* __restrict__ W, const float* __restrict__ ctx2,
                   __bf16* __restrict__ srcT) {
    const int b = blockIdx.y;
    const int g = blockIdx.x * 256 + threadIdx.x;   // 16384 = 256 d * 64 l
    const int d = g >> 6;
    const int l = g & 63;
    const float* wr = W + (size_t)d * CDF;
    const float* cr = ctx2 + (size_t)b * CDF * LL + l;
    float acc = 0.f;
    for (int c = 0; c < CDF; ++c) acc += wr[c] * cr[(size_t)c * LL];
    srcT[((size_t)b * DF + d) * LL + l] = (__bf16)acc;
}

// ---------------------------------------------------------------------------
// K3: fused main kernel. One workgroup = (b, 64-row q block), 128 thr = 4 waves.
//   Stage:   input[b, :, qblock] f32 -> LDS bf16 tile [64 q][256 d] (transposed)
//   Phase 1: per wave, attn[16q x 64L] via WMMA (K=256, A from LDS), norm_t
//            from A frags (4 independent FMA chains), cosine-norm +
//            mask(q%32 quirk) + softmax (probs kept in accumulator regs),
//            barrier, attn -> LDS (aliases tile).
//   Phase 2: attn_out transposed write; wc[64d slice, 64q] via WMMA (K=64,
//            B from LDS attn with f32->bf16 cvt).
// ---------------------------------------------------------------------------
__global__ __launch_bounds__(128)
void k_bu_attn_main(const float*  __restrict__ input,   // [B,DF,QK]
                    const __bf16* __restrict__ ctx1T,   // [B,LL,DF]
                    const __bf16* __restrict__ srcT,    // [B,DF,LL]
                    const float*  __restrict__ norm_c,  // [B,LL]
                    const int*    __restrict__ mask,    // [B,LL]
                    float* __restrict__ out_wc,         // [B,DF,QK]
                    float* __restrict__ out_attn) {     // [B,LL,QK]
    // tile: 64*264*2 = 33792 B ; attn: 64*68*4 = 17408 B  (aliased)
    __shared__ __attribute__((aligned(16))) char smem[64 * TROW * 2];
    __bf16* tile   = (__bf16*)smem;
    float*  attn_s = (float*)smem;

    const int tid  = threadIdx.x;
    const int w    = tid >> 5;        // wave 0..3
    const int lane = tid & 31;
    const int n    = lane & 15;
    const int hi   = lane >> 4;
    const int b    = blockIdx.y;
    const int qbase = blockIdx.x * 64;
    const int q0   = qbase + w * 16;

    // ---------------- Stage: transpose input slice into LDS (f32 -> bf16) ----
    {
        const float* ibase = input + (size_t)b * DF * QK + qbase;
        const int qloc = tid & 63;
        const int dph  = tid >> 6;                 // 0..1
#pragma unroll 4
        for (int it = 0; it < 64; ++it) {
            const int dp = dph + it * 2;           // dword index in d: 0..127
            const float x0 = ibase[(size_t)(2 * dp)     * QK + qloc];
            const float x1 = ibase[(size_t)(2 * dp + 1) * QK + qloc];
            union { __bf16 h[2]; unsigned u; } pk;
            pk.h[0] = (__bf16)x0;
            pk.h[1] = (__bf16)x1;
            *reinterpret_cast<unsigned*>(tile + qloc * TROW + 2 * dp) = pk.u;
        }
    }
    __syncthreads();

    // ---------------- Phase 1: attn = targetT @ ctx1 (WMMA, K=256) -----------
    v8f c0 = {0,0,0,0,0,0,0,0}, c1 = c0, c2 = c0, c3 = c0;
    float nta = 0.f, ntb = 0.f, ntc = 0.f, ntd = 0.f;   // 4 independent chains
    const __bf16* arow = tile + (size_t)(w * 16 + n) * TROW;
    const __bf16* brow = ctx1T + (size_t)b * LL * DF;

#pragma unroll
    for (int kk = 0; kk < 8; ++kk) {
        const int kb = kk * 32;
        const v8bf alo = *reinterpret_cast<const v8bf*>(arow + kb + 16 * hi);
        const v8bf ahi = *reinterpret_cast<const v8bf*>(arow + kb + 16 * hi + 8);
        const v16bf a = cat8(alo, ahi);
#pragma unroll
        for (int i = 0; i < 8; i += 4) {
            const float x0 = (float)alo[i + 0]; nta += x0 * x0;
            const float x1 = (float)alo[i + 1]; ntb += x1 * x1;
            const float x2 = (float)alo[i + 2]; ntc += x2 * x2;
            const float x3 = (float)alo[i + 3]; ntd += x3 * x3;
            const float y0 = (float)ahi[i + 0]; nta += y0 * y0;
            const float y1 = (float)ahi[i + 1]; ntb += y1 * y1;
            const float y2 = (float)ahi[i + 2]; ntc += y2 * y2;
            const float y3 = (float)ahi[i + 3]; ntd += y3 * y3;
        }
        const v16bf b0 = *reinterpret_cast<const v16bf*>(brow + (size_t)(0 * 16 + n) * DF + kb + 16 * hi);
        const v16bf b1 = *reinterpret_cast<const v16bf*>(brow + (size_t)(1 * 16 + n) * DF + kb + 16 * hi);
        const v16bf b2 = *reinterpret_cast<const v16bf*>(brow + (size_t)(2 * 16 + n) * DF + kb + 16 * hi);
        const v16bf b3 = *reinterpret_cast<const v16bf*>(brow + (size_t)(3 * 16 + n) * DF + kb + 16 * hi);
        c0 = wmma_bf16(a, b0, c0);
        c1 = wmma_bf16(a, b1, c1);
        c2 = wmma_bf16(a, b2, c2);
        c3 = wmma_bf16(a, b3, c3);
    }

    // norm_t: lane pair (m, m+16) holds disjoint K halves of row m.
    float nt2 = (nta + ntb) + (ntc + ntd);
    nt2 += __shfl_xor(nt2, 16, 32);
    const float ntm = sqrtf(nt2);              // lane s: norm of row s&15
    float nt_row[8];
#pragma unroll
    for (int r = 0; r < 8; ++r) nt_row[r] = __shfl(ntm, r + 8 * hi, 32);

    const float nc0 = norm_c[b * LL + 0 * 16 + n];
    const float nc1 = norm_c[b * LL + 1 * 16 + n];
    const float nc2 = norm_c[b * LL + 2 * 16 + n];
    const float nc3 = norm_c[b * LL + 3 * 16 + n];

#pragma unroll
    for (int r = 0; r < 8; ++r) {
        const int m = r + 8 * hi;                 // row within wave tile
        const int mrow = (q0 + m) & 31;           // reference's tile() quirk
        float v[4] = { c0[r], c1[r], c2[r], c3[r] };
        const float nc[4] = { nc0, nc1, nc2, nc3 };
#pragma unroll
        for (int t = 0; t < 4; ++t) {
            const float den = fmaxf(nt_row[r] * nc[t], 1e-8f);
            float val = v[t] / den;
            if (mask[mrow * LL + t * 16 + n]) val = -1e30f;
            v[t] = val;
        }
        float mx = fmaxf(fmaxf(v[0], v[1]), fmaxf(v[2], v[3]));
#pragma unroll
        for (int off = 8; off >= 1; off >>= 1) mx = fmaxf(mx, __shfl_xor(mx, off, 32));
        float s = 0.f;
#pragma unroll
        for (int t = 0; t < 4; ++t) { v[t] = __expf(v[t] - mx); s += v[t]; }
#pragma unroll
        for (int off = 8; off >= 1; off >>= 1) s += __shfl_xor(s, off, 32);
        const float inv = 1.0f / s;
        // keep probabilities in the accumulator registers (LDS still holds tile)
        c0[r] = v[0] * inv;
        c1[r] = v[1] * inv;
        c2[r] = v[2] * inv;
        c3[r] = v[3] * inv;
    }

    __syncthreads();   // all waves done reading the input tile

    // attn -> LDS (overwrites the aliased tile region)
#pragma unroll
    for (int r = 0; r < 8; ++r) {
        const int m = r + 8 * hi;
        float* rowp = attn_s + (size_t)(w * 16 + m) * AROW + n;
        rowp[0 * 16] = c0[r];
        rowp[1 * 16] = c1[r];
        rowp[2 * 16] = c2[r];
        rowp[3 * 16] = c3[r];
    }
    __syncthreads();

    // ---------------- attn_out (transposed, coalesced over q) ----------------
    float* oat = out_attn + (size_t)b * LL * QK + qbase;
    for (int idx = tid; idx < LL * 64; idx += 128) {
        const int l  = idx >> 6;
        const int ql = idx & 63;
        oat[(size_t)l * QK + ql] = attn_s[ql * AROW + l];
    }

    // ---------------- Phase 2: wc[d, q] = sum_l srcT[d,l] * attn[q,l] --------
    v8f acc[4][4];
    const v8f vz = {0,0,0,0,0,0,0,0};
#pragma unroll
    for (int dt = 0; dt < 4; ++dt)
#pragma unroll
        for (int qt = 0; qt < 4; ++qt) acc[dt][qt] = vz;

    const __bf16* srow = srcT + ((size_t)b * DF + w * 64) * LL;
#pragma unroll
    for (int ks = 0; ks < 2; ++ks) {
        const int kb = ks * 32;
        v16bf af[4];
#pragma unroll
        for (int dt = 0; dt < 4; ++dt) {
            const v8bf lo = *reinterpret_cast<const v8bf*>(srow + (size_t)(dt * 16 + n) * LL + kb + 16 * hi);
            const v8bf hiw = *reinterpret_cast<const v8bf*>(srow + (size_t)(dt * 16 + n) * LL + kb + 16 * hi + 8);
            af[dt] = cat8(lo, hiw);
        }
#pragma unroll
        for (int qt = 0; qt < 4; ++qt) {
            const float* lrow = attn_s + (size_t)(qt * 16 + n) * AROW + kb + 16 * hi;
            const float4 f0 = *reinterpret_cast<const float4*>(lrow + 0);
            const float4 f1 = *reinterpret_cast<const float4*>(lrow + 4);
            const float4 f2 = *reinterpret_cast<const float4*>(lrow + 8);
            const float4 f3 = *reinterpret_cast<const float4*>(lrow + 12);
            v16bf bb;
            bb[0]  = (__bf16)f0.x; bb[1]  = (__bf16)f0.y; bb[2]  = (__bf16)f0.z; bb[3]  = (__bf16)f0.w;
            bb[4]  = (__bf16)f1.x; bb[5]  = (__bf16)f1.y; bb[6]  = (__bf16)f1.z; bb[7]  = (__bf16)f1.w;
            bb[8]  = (__bf16)f2.x; bb[9]  = (__bf16)f2.y; bb[10] = (__bf16)f2.z; bb[11] = (__bf16)f2.w;
            bb[12] = (__bf16)f3.x; bb[13] = (__bf16)f3.y; bb[14] = (__bf16)f3.z; bb[15] = (__bf16)f3.w;
#pragma unroll
            for (int dt = 0; dt < 4; ++dt)
                acc[dt][qt] = wmma_bf16(af[dt], bb, acc[dt][qt]);
        }
    }

    float* owc = out_wc + (size_t)b * DF * QK;
#pragma unroll
    for (int dt = 0; dt < 4; ++dt)
#pragma unroll
        for (int qt = 0; qt < 4; ++qt)
#pragma unroll
            for (int r = 0; r < 8; ++r) {
                const int d = w * 64 + dt * 16 + r + 8 * hi;
                const int q = qbase + qt * 16 + n;
                owc[(size_t)d * QK + q] = acc[dt][qt][r];
            }
}

// ---------------------------------------------------------------------------
extern "C" void kernel_launch(void* const* d_in, const int* in_sizes, int n_in,
                              void* d_out, int out_size, void* d_ws, size_t ws_size,
                              hipStream_t stream) {
    (void)in_sizes; (void)n_in; (void)out_size; (void)ws_size;
    const float* input = (const float*)d_in[0];   // [B,DF,64,64]
    const float* ctx1  = (const float*)d_in[1];   // [B,DF,LL]
    const float* ctx2  = (const float*)d_in[2];   // [B,CDF,LL]
    const int*   mask  = (const int*)  d_in[3];   // [B,LL] bool->int
    const float* W     = (const float*)d_in[4];   // [DF,CDF]

    float* out = (float*)d_out;
    float* out_wc   = out;                                   // B*DF*QK floats
    float* out_attn = out + (size_t)BB * DF * QK;            // B*LL*QK floats

    char* ws = (char*)d_ws;
    __bf16* ctx1T  = (__bf16*)ws;                            // B*LL*DF*2 = 1 MiB
    __bf16* srcT   = (__bf16*)(ws + (size_t)1048576);        // B*DF*LL*2 = 1 MiB
    float*  norm_c = (float*) (ws + (size_t)2097152);        // B*LL*4    = 8 KiB

    k_prep_ctx<<<dim3(32), 256, 0, stream>>>(ctx1, ctx1T, norm_c);
    k_prep_source<<<dim3(64, 32), 256, 0, stream>>>(W, ctx2, srcT);
    k_bu_attn_main<<<dim3(64, 32), 128, 0, stream>>>(input, ctx1T, srcT, norm_c,
                                                     mask, out_wc, out_attn);
}
